// MultimodalTransitionModel_64871186038796
// MI455X (gfx1250) — matrix operand order repaired and checked by reading, not compiled, not run on hardware
//
#include <hip/hip_runtime.h>
#include <hip/hip_bf16.h>

// ---------------- problem constants ----------------
#define BB   512
#define TT   50
#define BEL  1024
#define SS   256
#define AA   32
#define HID  1024
#define EI   1024
#define ES   256

typedef __attribute__((ext_vector_type(16))) _Float16 v16h;
typedef __attribute__((ext_vector_type(8)))  _Float16 h8;
typedef __attribute__((ext_vector_type(8)))  float    v8f;

// ---------------- f32 -> f16 weight conversion ----------------
__global__ void cvt_f16_kernel(const float* __restrict__ src,
                               _Float16* __restrict__ dst, int n) {
  for (int i = blockIdx.x * blockDim.x + threadIdx.x; i < n;
       i += gridDim.x * blockDim.x) {
    dst[i] = (_Float16)src[i];
  }
}

// ---------------- WMMA GEMM: C = act( [A0|A1] @ W^T + bias ) ----------------
// Block tile 128x128x32, 8 wave32 waves, wave tile 32x64 (2x4 WMMA tiles).
// Double-buffered LDS; next-tile global loads issued before the WMMA block.
// Template-specialized on concat (A1 present) and per-row scale (A0 part only)
// so the common path has a branch-free inner loop.
template <bool CONCAT, bool SCALE>
__global__ __launch_bounds__(256) void gemm_wmma_db(
    const float* __restrict__ A0, int K0,
    const float* __restrict__ A1, int K1,
    const _Float16* __restrict__ W,
    const float* __restrict__ bias,
    const float* __restrict__ rowScale,
    float* __restrict__ C, int N, int act)
{
  const int K = CONCAT ? (K0 + K1) : K0;
  __shared__ _Float16 As[2][128][40];   // 2 x 128x32 tile, padded stride
  __shared__ _Float16 Bs[2][128][40];   // 2 x 128x32 tile of W rows

  const int tid  = threadIdx.x;
  const int wave = tid >> 5;
  const int lane = tid & 31;
  const int wm   = (wave & 3) * 32;     // 4 row-waves
  const int wn   = (wave >> 2) * 64;    // 2 col-waves x 64
  const int m0   = blockIdx.y * 128;
  const int n0   = blockIdx.x * 128;

  // A staging geometry: 4 float4 per thread
  const int arow = tid >> 3;            // + i*32
  const int ac4  = (tid & 7) << 2;
  // B staging geometry: 2 h8 per thread
  // chunk c = tid + j*256: row = c>>2, col8 = (c&3)*8

  // hoisted per-row scales (loop-invariant)
  float rsc[4];
  if (SCALE) {
#pragma unroll
    for (int i = 0; i < 4; ++i) rsc[i] = rowScale[m0 + arow + i * 32];
  }

  float4 areg[4];
  h8     breg[2];
  int    nfromA0 = 1;

  auto loadA = [&](int k0) {
    const float* Asrc = A0; int lda = K0; int kk = k0; nfromA0 = 1;
    if (CONCAT && k0 >= K0) { Asrc = A1; lda = K1; kk = k0 - K0; nfromA0 = 0; }
#pragma unroll
    for (int i = 0; i < 4; ++i)
      areg[i] = *(const float4*)(Asrc + (long)(m0 + arow + i * 32) * lda + kk + ac4);
  };
  auto loadB = [&](int k0) {
#pragma unroll
    for (int j = 0; j < 2; ++j) {
      int c = tid + j * 256;
      int row = c >> 2, c8 = (c & 3) << 3;
      breg[j] = *(const h8*)(W + (long)(n0 + row) * K + k0 + c8);
    }
  };
  auto storeAB = [&](int buf) {
#pragma unroll
    for (int i = 0; i < 4; ++i) {
      float s = (SCALE && nfromA0) ? rsc[i] : 1.0f;
      int row = arow + i * 32;
      As[buf][row][ac4 + 0] = (_Float16)(areg[i].x * s);
      As[buf][row][ac4 + 1] = (_Float16)(areg[i].y * s);
      As[buf][row][ac4 + 2] = (_Float16)(areg[i].z * s);
      As[buf][row][ac4 + 3] = (_Float16)(areg[i].w * s);
    }
#pragma unroll
    for (int j = 0; j < 2; ++j) {
      int c = tid + j * 256;
      int row = c >> 2, c8 = (c & 3) << 3;
      *(h8*)(&Bs[buf][row][c8]) = breg[j];
    }
  };

  v8f acc[2][4] = {};

  // prologue: stage tile 0 into buffer 0
  loadA(0); loadB(0);
  storeAB(0);
  __syncthreads();

  const int nkt = K >> 5;
  const int l15  = lane & 15;
  const int ksel = (lane < 16) ? 0 : 8;   // A frag: lanes<16 K{0..7,16..23}, else K{8..15,24..31}
  const int bk   = (lane < 16) ? 0 : 8;   // B frag: lanes<16 K{0..7,16..23}, else K{8..15,24..31}

  for (int kt = 0; kt < nkt; ++kt) {
    const int cur = kt & 1;

    // issue next tile's global loads early (hide behind WMMA)
    if (kt + 1 < nkt) { loadA((kt + 1) << 5); loadB((kt + 1) << 5); }

    // ---- fragments per CDNA5 16-bit WMMA VGPR layouts ----
    v16h af[2], bf[4];
#pragma unroll
    for (int tm = 0; tm < 2; ++tm) {
      int mr = wm + tm * 16 + l15;
      h8 lo = *(const h8*)(&As[cur][mr][ksel]);
      h8 hi = *(const h8*)(&As[cur][mr][ksel + 16]);
      af[tm] = __builtin_shufflevector(lo, hi, 0,1,2,3,4,5,6,7,8,9,10,11,12,13,14,15);
    }
#pragma unroll
    for (int tn = 0; tn < 4; ++tn) {
      int nr = wn + tn * 16 + l15;          // one output column (W row) per lane
      h8 b0 = *(const h8*)(&Bs[cur][nr][bk]);
      h8 b1 = *(const h8*)(&Bs[cur][nr][bk + 16]);
      bf[tn] = __builtin_shufflevector(b0, b1, 0,1,2,3,4,5,6,7,8,9,10,11,12,13,14,15);
    }

#pragma unroll
    for (int tm = 0; tm < 2; ++tm)
#pragma unroll
      for (int tn = 0; tn < 4; ++tn)
        acc[tm][tn] = __builtin_amdgcn_wmma_f32_16x16x32_f16(
            false, af[tm], false, bf[tn], (short)0, acc[tm][tn], false, false);

    if (kt + 1 < nkt) storeAB(cur ^ 1);
    __syncthreads();
  }

  // ---- epilogue: bias + optional ReLU, per documented C layout ----
  const int hs = (lane >> 4) * 8;   // lanes 16-31 hold M rows 8..15
#pragma unroll
  for (int tm = 0; tm < 2; ++tm) {
#pragma unroll
    for (int tn = 0; tn < 4; ++tn) {
      int colg = n0 + wn + tn * 16 + l15;
      float bv = bias ? bias[colg] : 0.0f;
#pragma unroll
      for (int r = 0; r < 8; ++r) {
        int rowg = m0 + wm + tm * 16 + hs + r;
        float v = acc[tm][tn][r] + bv;
        if (act) v = fmaxf(v, 0.0f);
        C[(long)rowg * N + colg] = v;
      }
    }
  }
}

// ---------------- GRU gate fusion (in-place belief update) ----------------
__global__ void gru_fuse_kernel(const float* __restrict__ gi,
                                const float* __restrict__ gh,
                                float* __restrict__ h,
                                float* __restrict__ beliefs_out) {
  int idx = blockIdx.x * blockDim.x + threadIdx.x;   // B*BEL threads
  int b = idx >> 10, d = idx & 1023;
  const int D = BEL;
  long base = (long)b * 3 * D;
  float ir = gi[base + d], iz = gi[base + D + d], in_ = gi[base + 2 * D + d];
  float hr = gh[base + d], hz = gh[base + D + d], hn  = gh[base + 2 * D + d];
  float r = 1.0f / (1.0f + __expf(-(ir + hr)));
  float z = 1.0f / (1.0f + __expf(-(iz + hz)));
  float n = tanhf(in_ + r * hn);
  float out = (1.0f - z) * n + z * h[idx];
  h[idx] = out;
  beliefs_out[idx] = out;
}

// ---------------- PoE fusion + output writes ----------------
__device__ __forceinline__ float softplusf(float x) {
  return (x > 20.0f) ? x : log1pf(__expf(x));
}

__global__ void poe_fuse_kernel(const float* __restrict__ lsp,
                                const float* __restrict__ lsi,
                                const float* __restrict__ lss,
                                float* __restrict__ post,
                                float* __restrict__ o_ps, float* __restrict__ o_pm,
                                float* __restrict__ o_pstd,
                                float* __restrict__ o_qs, float* __restrict__ o_qm,
                                float* __restrict__ o_qstd) {
  int idx = blockIdx.x * blockDim.x + threadIdx.x;   // B*S threads
  int b = idx >> 8, d = idx & 255;
  long base = (long)b * (2 * SS);
  float pl  = lsp[base + d];
  float psc = softplusf(lsp[base + SS + d]) + 0.1f;
  float il  = lsi[base + d];
  float isc = softplusf(lsi[base + SS + d]) + 0.1f;
  float sl  = lss[base + d];
  float ssc = softplusf(lss[base + SS + d]) + 0.1f;
  float pi_ = 1.0f / (isc * isc);
  float ps_ = 1.0f / (ssc * ssc);
  float var = 1.0f / (1.0f + pi_ + ps_);   // + unit-Gaussian expert
  float ql  = var * (il * pi_ + sl * ps_);
  float qsd = sqrtf(var);
  post[idx] = ql;
  o_ps[idx] = pl;  o_pm[idx] = pl;  o_pstd[idx] = psc;
  o_qs[idx] = ql;  o_qm[idx] = ql;  o_qstd[idx] = qsd;
}

// ---------------- host launcher ----------------
static inline size_t align256(size_t x) { return (x + 255) & ~(size_t)255; }

extern "C" void kernel_launch(void* const* d_in, const int* in_sizes, int n_in,
                              void* d_out, int out_size, void* d_ws, size_t ws_size,
                              hipStream_t stream) {
  const float* prev_state  = (const float*)d_in[0];   // [B,S]
  const float* actions     = (const float*)d_in[1];   // [T,B,A]
  const float* prev_belief = (const float*)d_in[2];   // [B,BEL]
  const float* obs_image   = (const float*)d_in[3];   // [T,B,EI]
  const float* obs_sound   = (const float*)d_in[4];   // [T,B,ES]
  const float* nonterm     = (const float*)d_in[5];   // [T,B,1]
  const float* W_emb = (const float*)d_in[6];  const float* b_emb = (const float*)d_in[7];
  const float* W_ih  = (const float*)d_in[8];  const float* b_ih  = (const float*)d_in[9];
  const float* W_hh  = (const float*)d_in[10]; const float* b_hh  = (const float*)d_in[11];
  const float* W_s1  = (const float*)d_in[12]; const float* b_s1  = (const float*)d_in[13];
  const float* W_s2  = (const float*)d_in[14]; const float* b_s2  = (const float*)d_in[15];
  const float* W_ei1 = (const float*)d_in[16]; const float* b_ei1 = (const float*)d_in[17];
  const float* W_ei2 = (const float*)d_in[18]; const float* b_ei2 = (const float*)d_in[19];
  const float* W_es1 = (const float*)d_in[20]; const float* b_es1 = (const float*)d_in[21];
  const float* W_es2 = (const float*)d_in[22]; const float* b_es2 = (const float*)d_in[23];

  // ---- workspace layout ----
  char* ws = (char*)d_ws;
  size_t off = 0;
  auto takeH = [&](size_t elems) { _Float16* p = (_Float16*)(ws + off);
                                   off = align256(off + elems * sizeof(_Float16)); return p; };
  auto takeF = [&](size_t elems) { float* p = (float*)(ws + off);
                                   off = align256(off + elems * sizeof(float)); return p; };

  _Float16* hEmb = takeH((size_t)BEL * (SS + AA));
  _Float16* hIh  = takeH((size_t)3 * BEL * BEL);
  _Float16* hHh  = takeH((size_t)3 * BEL * BEL);
  _Float16* hS1  = takeH((size_t)HID * BEL);
  _Float16* hS2  = takeH((size_t)2 * SS * HID);
  _Float16* hEi1 = takeH((size_t)HID * (BEL + EI));
  _Float16* hEi2 = takeH((size_t)2 * SS * HID);
  _Float16* hEs1 = takeH((size_t)HID * (BEL + ES));
  _Float16* hEs2 = takeH((size_t)2 * SS * HID);

  float* belief = takeF((size_t)BB * BEL);
  float* post   = takeF((size_t)BB * SS);
  float* hidden = takeF((size_t)BB * BEL);
  float* gi     = takeF((size_t)BB * 3 * BEL);
  float* gh     = takeF((size_t)BB * 3 * BEL);
  float* hhead  = takeF((size_t)BB * HID);
  float* lsp    = takeF((size_t)BB * 2 * SS);
  float* lsi    = takeF((size_t)BB * 2 * SS);
  float* lss    = takeF((size_t)BB * 2 * SS);
  (void)ws_size; (void)in_sizes; (void)n_in; (void)out_size;

  // ---- output slices (tuple concatenated flat) ----
  float* out      = (float*)d_out;
  float* o_bel    = out;
  float* o_pstate = o_bel    + (long)TT * BB * BEL;
  float* o_pmean  = o_pstate + (long)TT * BB * SS;
  float* o_pstd   = o_pmean  + (long)TT * BB * SS;
  float* o_qstate = o_pstd   + (long)TT * BB * SS;
  float* o_qmean  = o_qstate + (long)TT * BB * SS;
  float* o_qstd   = o_qmean  + (long)TT * BB * SS;

  // ---- one-time weight conversion to f16 ----
  auto cvt = [&](const float* s, _Float16* d, int n) {
    cvt_f16_kernel<<<512, 256, 0, stream>>>(s, d, n);
  };
  cvt(W_emb, hEmb, BEL * (SS + AA));
  cvt(W_ih,  hIh,  3 * BEL * BEL);
  cvt(W_hh,  hHh,  3 * BEL * BEL);
  cvt(W_s1,  hS1,  HID * BEL);
  cvt(W_s2,  hS2,  2 * SS * HID);
  cvt(W_ei1, hEi1, HID * (BEL + EI));
  cvt(W_ei2, hEi2, 2 * SS * HID);
  cvt(W_es1, hEs1, HID * (BEL + ES));
  cvt(W_es2, hEs2, 2 * SS * HID);

  // ---- init recurrent state ----
  hipMemcpyAsync(belief, prev_belief, (size_t)BB * BEL * sizeof(float),
                 hipMemcpyDeviceToDevice, stream);
  hipMemcpyAsync(post, prev_state, (size_t)BB * SS * sizeof(float),
                 hipMemcpyDeviceToDevice, stream);

  auto gemm = [&](const float* A0, int K0, const float* A1, int K1,
                  const _Float16* W, const float* bias, const float* rs,
                  float* C, int N, int act) {
    dim3 grid(N / 128, BB / 128);
    if (rs)
      gemm_wmma_db<true, true><<<grid, 256, 0, stream>>>(A0, K0, A1, K1, W, bias, rs, C, N, act);
    else if (A1)
      gemm_wmma_db<true, false><<<grid, 256, 0, stream>>>(A0, K0, A1, K1, W, bias, rs, C, N, act);
    else
      gemm_wmma_db<false, false><<<grid, 256, 0, stream>>>(A0, K0, A1, K1, W, bias, rs, C, N, act);
  };

  for (int t = 0; t < TT; ++t) {
    const float* a_t = actions   + (long)t * BB * AA;
    const float* nt  = nonterm   + (long)t * BB;
    const float* oi  = obs_image + (long)t * BB * EI;
    const float* os  = obs_sound + (long)t * BB * ES;

    // hidden = relu([post*nt | a] @ W_emb^T + b_emb)
    gemm(post, SS, a_t, AA, hEmb, b_emb, nt, hidden, BEL, 1);
    // GRU gates
    gemm(hidden, BEL, nullptr, 0, hIh, b_ih, nullptr, gi, 3 * BEL, 0);
    gemm(belief, BEL, nullptr, 0, hHh, b_hh, nullptr, gh, 3 * BEL, 0);
    gru_fuse_kernel<<<(BB * BEL) / 256, 256, 0, stream>>>(
        gi, gh, belief, o_bel + (long)t * BB * BEL);

    // prior head
    gemm(belief, BEL, nullptr, 0, hS1, b_s1, nullptr, hhead, HID, 1);
    gemm(hhead, HID, nullptr, 0, hS2, b_s2, nullptr, lsp, 2 * SS, 0);
    // image head ([belief | obs_image])
    gemm(belief, BEL, oi, EI, hEi1, b_ei1, nullptr, hhead, HID, 1);
    gemm(hhead, HID, nullptr, 0, hEi2, b_ei2, nullptr, lsi, 2 * SS, 0);
    // sound head ([belief | obs_sound])
    gemm(belief, BEL, os, ES, hEs1, b_es1, nullptr, hhead, HID, 1);
    gemm(hhead, HID, nullptr, 0, hEs2, b_es2, nullptr, lss, 2 * SS, 0);

    // PoE fusion + all per-step outputs + next post
    poe_fuse_kernel<<<(BB * SS) / 256, 256, 0, stream>>>(
        lsp, lsi, lss, post,
        o_pstate + (long)t * BB * SS, o_pmean + (long)t * BB * SS,
        o_pstd   + (long)t * BB * SS,
        o_qstate + (long)t * BB * SS, o_qmean + (long)t * BB * SS,
        o_qstd   + (long)t * BB * SS);
  }
}
